// GCN_4243427689159
// MI455X (gfx1250) — matrix operand order repaired and checked by reading, not compiled, not run on hardware
//
#include <hip/hip_runtime.h>

typedef __attribute__((ext_vector_type(16))) _Float16 v16h;
typedef __attribute__((ext_vector_type(8)))  _Float16 v8h;
typedef __attribute__((ext_vector_type(8)))  float    v8f;

#define N_NODES   100000
#define N_EDGES   3200000
#define DIM       256
#define NCLS      40
#define ROWS_PB   64        // h rows per block = 4 WMMA m-tiles
#define LSTRIDE   264       // padded f16 row stride (264*2B = 528B -> conflict-free)

// ---------------------------------------------------------------------------
// Kernel 1: adj_row (sorted) -> CSR row_ptr via lower_bound
// ---------------------------------------------------------------------------
__global__ __launch_bounds__(256) void gcn_row_ptr(const int* __restrict__ adj_row,
                                                   int* __restrict__ row_ptr) {
    int r = blockIdx.x * blockDim.x + threadIdx.x;
    if (r > N_NODES) return;
    int lo = 0, hi = N_EDGES;
    while (lo < hi) {
        int mid = (lo + hi) >> 1;
        if (adj_row[mid] < r) lo = mid + 1; else hi = mid;
    }
    row_ptr[r] = lo;
}

// ---------------------------------------------------------------------------
// Kernel 2 (fused): h = relu(A·X)*mask*2 computed into LDS (f16), then
// z = h @ W^T + b via v_wmma_f32_16x16x32_f16. One wave per 16-row m-tile.
// ---------------------------------------------------------------------------
__global__ __launch_bounds__(128) void gcn_spmm_gemm(
    const float* __restrict__ x,
    const float* __restrict__ adj_vals,
    const int*   __restrict__ adj_col,
    const float* __restrict__ W,
    const float* __restrict__ bias,
    const float* __restrict__ drop_mask,
    const int*   __restrict__ row_ptr,
    float*       __restrict__ z)
{
    __shared__ _Float16 hT[ROWS_PB * LSTRIDE];   // 64 rows of h, f16, padded
    __shared__ _Float16 wT[48 * LSTRIDE];        // W padded to 48 rows, f16

    const int tid     = threadIdx.x;
    const int lane    = tid & 31;
    const int wave    = tid >> 5;                // 0..3
    const int rowBase = blockIdx.x * ROWS_PB;

    // ---- stage W (40x256 f32) -> LDS f16, rows 40..47 zero ----
    for (int i = tid; i < 48 * DIM; i += 128) {
        int r = i >> 8, c = i & (DIM - 1);
        float v = (r < NCLS) ? W[r * DIM + c] : 0.0f;
        wT[r * LSTRIDE + c] = (_Float16)v;
    }

    // ---- phase 1: wave w computes its own 16 h-rows into LDS ----
    for (int i = 0; i < 16; ++i) {
        const int lrow = wave * 16 + i;
        const int row  = rowBase + lrow;
        float acc[8] = {0.0f, 0.0f, 0.0f, 0.0f, 0.0f, 0.0f, 0.0f, 0.0f};
        if (row < N_NODES) {
            const int e1 = row_ptr[row + 1];
            #pragma unroll 2
            for (int e = row_ptr[row]; e < e1; ++e) {
                const int   col = adj_col[e];
                const float val = adj_vals[e];
                const float4* xr = (const float4*)(x + (size_t)col * DIM + lane * 8);
                float4 a = xr[0];
                float4 b = xr[1];
                acc[0] = fmaf(val, a.x, acc[0]); acc[1] = fmaf(val, a.y, acc[1]);
                acc[2] = fmaf(val, a.z, acc[2]); acc[3] = fmaf(val, a.w, acc[3]);
                acc[4] = fmaf(val, b.x, acc[4]); acc[5] = fmaf(val, b.y, acc[5]);
                acc[6] = fmaf(val, b.z, acc[6]); acc[7] = fmaf(val, b.w, acc[7]);
            }
            const float4* dm = (const float4*)(drop_mask + (size_t)row * DIM + lane * 8);
            float4 m0 = dm[0], m1 = dm[1];
            float mm[8] = {m0.x, m0.y, m0.z, m0.w, m1.x, m1.y, m1.z, m1.w};
            #pragma unroll
            for (int j = 0; j < 8; ++j)
                acc[j] = fmaxf(acc[j], 0.0f) * mm[j] * 2.0f;   // relu + dropout/KEEP
        }
        v8h hv;
        #pragma unroll
        for (int j = 0; j < 8; ++j) hv[j] = (_Float16)acc[j];
        *(v8h*)&hT[lrow * LSTRIDE + lane * 8] = hv;            // ds_store_b128
    }
    __syncthreads();

    // ---- phase 2: WMMA 16x48 projection for this wave's m-tile ----
    const int m0 = rowBase + wave * 16;
    if (m0 < N_NODES) {                 // wave-uniform -> EXEC all-1s inside
        v8f accs[3] = {};
        const int nc   = lane & 15;
        const int half = lane >> 4;     // K-half select per ISA layout
        const int arow = wave * 16 + nc;
        #pragma unroll
        for (int ks = 0; ks < 8; ++ks) {
            const int kb = ks * 32;
            // A fragment: 16x32 f16. lanes 0-15: K kb+0..7 / kb+16..23,
            // lanes 16-31: K kb+8..15 / kb+24..31  (ISA 7.12.2)
            const _Float16* ap = &hT[arow * LSTRIDE + kb + half * 8];
            v8h a0 = *(const v8h*)(ap);
            v8h a1 = *(const v8h*)(ap + 16);
            v16h A;
            #pragma unroll
            for (int j = 0; j < 8; ++j) { A[j] = a0[j]; A[8 + j] = a1[j]; }
            #pragma unroll
            for (int ct = 0; ct < 3; ++ct) {
                // B fragment: 32x16 f16. lanes 0-15: K kb+0..7 / kb+16..23; 16-31: K kb+8..15 / kb+24..31
                const _Float16* bp = &wT[(ct * 16 + nc) * LSTRIDE + kb + half * 8];
                v8h b0 = *(const v8h*)(bp);
                v8h b1 = *(const v8h*)(bp + 16);
                v16h B;
                #pragma unroll
                for (int j = 0; j < 8; ++j) { B[j] = b0[j]; B[8 + j] = b1[j]; }
                accs[ct] = __builtin_amdgcn_wmma_f32_16x16x32_f16(
                    false, A, false, B, (short)0, accs[ct], false, false);
            }
        }
        // store z = acc + bias. C/D layout: VGPR r -> M = half*8 + r, N = nc
        const int mtop = half * 8;
        #pragma unroll
        for (int ct = 0; ct < 3; ++ct) {
            const int ncol = ct * 16 + nc;
            if (ncol < NCLS) {
                const float bb = bias[ncol];
                #pragma unroll
                for (int r = 0; r < 8; ++r)
                    z[(size_t)(m0 + mtop + r) * NCLS + ncol] = accs[ct][r] + bb;
            }
        }
    }
}

// ---------------------------------------------------------------------------
// Kernel 3: out = A · z   (wave per row; lane covers class lane and lane+32)
// ---------------------------------------------------------------------------
__global__ __launch_bounds__(256) void gcn_spmm_out(
    const float* __restrict__ z,
    const float* __restrict__ adj_vals,
    const int*   __restrict__ adj_col,
    const int*   __restrict__ row_ptr,
    float*       __restrict__ out)
{
    const int lane = threadIdx.x & 31;
    const int row  = blockIdx.x * 8 + (threadIdx.x >> 5);
    if (row >= N_NODES) return;
    const int e1 = row_ptr[row + 1];
    float a0 = 0.0f, a1 = 0.0f;
    #pragma unroll 2
    for (int e = row_ptr[row]; e < e1; ++e) {
        const int   col = adj_col[e];
        const float val = adj_vals[e];
        const float* zr = z + (size_t)col * NCLS;
        a0 = fmaf(val, zr[lane], a0);
        a1 = fmaf(val, zr[32 + lane], a1);   // z padded: always in-bounds
    }
    out[(size_t)row * NCLS + lane] = a0;
    if (lane < 8) out[(size_t)row * NCLS + 32 + lane] = a1;
}

// ---------------------------------------------------------------------------
extern "C" void kernel_launch(void* const* d_in, const int* in_sizes, int n_in,
                              void* d_out, int out_size, void* d_ws, size_t ws_size,
                              hipStream_t stream)
{
    (void)in_sizes; (void)n_in; (void)out_size; (void)ws_size;
    const float* x         = (const float*)d_in[0];
    const float* adj_vals  = (const float*)d_in[1];
    const float* W         = (const float*)d_in[2];
    const float* bias      = (const float*)d_in[3];
    const float* drop_mask = (const float*)d_in[4];
    const int*   adj_row   = (const int*)d_in[5];
    const int*   adj_col   = (const int*)d_in[6];
    float* out = (float*)d_out;

    // workspace: [row_ptr : (N+1) ints][pad to 256B][z : N*40 + 64 floats]
    int* row_ptr = (int*)d_ws;
    size_t zoff  = (((size_t)(N_NODES + 1) * sizeof(int)) + 255) & ~(size_t)255;
    float* z     = (float*)((char*)d_ws + zoff);

    gcn_row_ptr<<<dim3((N_NODES + 1 + 255) / 256), dim3(256), 0, stream>>>(adj_row, row_ptr);
    gcn_spmm_gemm<<<dim3((N_NODES + ROWS_PB - 1) / ROWS_PB), dim3(128), 0, stream>>>(
        x, adj_vals, adj_col, W, bias, drop_mask, row_ptr, z);
    gcn_spmm_out<<<dim3((N_NODES + 7) / 8), dim3(256), 0, stream>>>(
        z, adj_vals, adj_col, row_ptr, out);
}
